// TopicEncoder_34024730919221
// MI455X (gfx1250) — matrix operand order repaired and checked by reading, not compiled, not run on hardware
//
#include <hip/hip_runtime.h>

// Problem constants (match reference)
#define B_ 32
#define S_ 512
#define H_ 1024
#define T_ 100
#define F_ 512
#define L_ 5
#define EPS_ 1e-5f

// GEMM tiling: 32 token-rows per block, 16 waves x 64 columns = N=1024 per block.
#define WAVES_G 16
#define TPB_G   (WAVES_G * 32)
#define NT_G    4              // 4 16-col N-tiles per wave

typedef __attribute__((ext_vector_type(16))) __bf16 v16bf;
typedef __attribute__((ext_vector_type(8)))  float  v8f;
typedef __attribute__((ext_vector_type(4)))  unsigned int u32x4;

union Frag16 { u32x4 q[2]; v16bf v; };

__device__ __forceinline__ unsigned short f2bf(float f) {
  unsigned int u = __float_as_uint(f);
  u += 0x7FFFu + ((u >> 16) & 1u);   // round-to-nearest-even truncation
  return (unsigned short)(u >> 16);
}

// ---------------- elementwise conversion f32 -> bf16, 4 elems/thread ----------------
__global__ __launch_bounds__(256) void cvt_f32_to_bf16_v4(
    const float* __restrict__ src, unsigned short* __restrict__ dst, int n4) {
  int i = blockIdx.x * 256 + threadIdx.x;
  if (i >= n4) return;
  float4 v = reinterpret_cast<const float4*>(src)[i];
  uint2 o;
  o.x = (unsigned)f2bf(v.x) | ((unsigned)f2bf(v.y) << 16);
  o.y = (unsigned)f2bf(v.z) | ((unsigned)f2bf(v.w) << 16);
  reinterpret_cast<uint2*>(dst)[i] = o;
}

__global__ __launch_bounds__(256) void zero_f32(float* __restrict__ p, int n) {
  int i = blockIdx.x * 256 + threadIdx.x;
  if (i < n) p[i] = 0.0f;
}

// per-batch sum of attention mask (S_=512, 256 threads, 2 elems/thread)
__global__ __launch_bounds__(256) void mask_rowsum(const float* __restrict__ mask,
                                                   float* __restrict__ msum) {
  __shared__ float red[8];
  int b = blockIdx.x, tid = threadIdx.x;
  float s = mask[b * S_ + tid] + mask[b * S_ + 256 + tid];
  #pragma unroll
  for (int off = 16; off > 0; off >>= 1) s += __shfl_down(s, off, 32);
  if ((tid & 31) == 0) red[tid >> 5] = s;
  __syncthreads();
  if (tid == 0) {
    float t = 0.f;
    #pragma unroll
    for (int w = 0; w < 8; ++w) t += red[w];
    msum[b] = t;
  }
}

// ---------------- fragment loads ----------------
// A lane layout (16-bit A 16x32): lane l -> row l&15; K octet base (l>>4)*8 and 16+(l>>4)*8.
// B lane layout (16-bit B 32x16): lane l -> column l&15; K octet base (l>>4)*8 and 16+(l>>4)*8.
__device__ __forceinline__ void loadA2(const unsigned short* __restrict__ Arow0,
                                       const unsigned short* __restrict__ Arow1,
                                       int k, int hi, Frag16& a0, Frag16& a1) {
  const unsigned short* ap0 = Arow0 + k + hi * 8;
  const unsigned short* ap1 = Arow1 + k + hi * 8;
  a0.q[0] = *reinterpret_cast<const u32x4*>(ap0);
  a0.q[1] = *reinterpret_cast<const u32x4*>(ap0 + 16);
  a1.q[0] = *reinterpret_cast<const u32x4*>(ap1);
  a1.q[1] = *reinterpret_cast<const u32x4*>(ap1 + 16);
}

__device__ __forceinline__ void loadB(const unsigned short* __restrict__ bp, int nt, Frag16& b) {
  unsigned short t[16];
  #pragma unroll
  for (int s = 0; s < 16; ++s) t[s] = bp[nt * 16 + (size_t)(s < 8 ? s : s + 8) * H_];
  __builtin_memcpy(&b, t, sizeof(t));
}

#define WMMA_BF16(A, Bv, C) \
  __builtin_amdgcn_wmma_f32_16x16x32_bf16(false, (A).v, false, (Bv).v, (short)0, (C), false, false)

// ---------------- shared WMMA mainloop (software pipelined) ----------------
// C0[16 x 64] += A0[16 x 1024] * B[1024 x 64]
// C1[16 x 64] += A1[16 x 1024] * B[1024 x 64]   (B fragments reused for both row tiles)
__device__ __forceinline__ void wmma_panel2(
    const unsigned short* __restrict__ Arow0,  // row (m0      + (lane&15)), element 0
    const unsigned short* __restrict__ Arow1,  // row (m0 + 16 + (lane&15)), element 0
    const unsigned short* __restrict__ Bbase,  // [H_,H_] row-major bf16 weight
    int n0, int lane, v8f acc0[NT_G], v8f acc1[NT_G]) {
  const int hi = lane >> 4;
  Frag16 a0c, a1c;
  loadA2(Arow0, Arow1, 0, hi, a0c, a1c);     // prologue: A for k=0
  for (int k = 0; k < H_; k += 32) {
    const unsigned short* bp = Bbase + (size_t)(k + hi * 8) * H_ + n0 + (lane & 15);
    // issue all four B fragments (64 x 16-bit loads) before any consumption
    Frag16 b0, b1, b2, b3;
    loadB(bp, 0, b0);
    loadB(bp, 1, b1);
    loadB(bp, 2, b2);
    loadB(bp, 3, b3);
    // issue next k-step's A fragments + prefetch before the WMMA block
    Frag16 a0n, a1n;
    if (k + 32 < H_) {
      __builtin_prefetch(bp + (size_t)32 * H_, 0, 1);  // next k-step's B row
      loadA2(Arow0, Arow1, k + 32, hi, a0n, a1n);
    } else {
      a0n = a0c; a1n = a1c;
    }
    // consume in load-issue order: each wait leaves later loads in flight
    acc0[0] = WMMA_BF16(a0c, b0, acc0[0]);
    acc1[0] = WMMA_BF16(a1c, b0, acc1[0]);
    acc0[1] = WMMA_BF16(a0c, b1, acc0[1]);
    acc1[1] = WMMA_BF16(a1c, b1, acc1[1]);
    acc0[2] = WMMA_BF16(a0c, b2, acc0[2]);
    acc1[2] = WMMA_BF16(a1c, b2, acc1[2]);
    acc0[3] = WMMA_BF16(a0c, b3, acc0[3]);
    acc1[3] = WMMA_BF16(a1c, b3, acc1[3]);
    a0c = a0n; a1c = a1n;
  }
}

// ---------------- adapter layer 1: h1 = relu(LN(seq @ W1[lang] + b1[lang])) -> bf16 ----------------
__global__ __launch_bounds__(TPB_G) void adapter_gemm1(
    const unsigned short* __restrict__ seqbf,   // [B_*S_, H_]
    const unsigned short* __restrict__ W1bf,    // [L_, H_, H_]
    const float* __restrict__ b1, const float* __restrict__ g1,
    const float* __restrict__ beta1, const int* __restrict__ lang_ids,
    unsigned short* __restrict__ h1bf) {
  const int tid = threadIdx.x, lane = tid & 31, wave = tid >> 5;
  const int hi = lane >> 4, lo = lane & 15;
  const int m0 = blockIdx.x * 32;          // 32 tokens per block, within one batch (512%32==0)
  const int b = m0 >> 9;                   // / S_
  const int lang = lang_ids[b];
  const int n0 = wave * (NT_G * 16);       // 64 columns per wave

  v8f acc0[NT_G], acc1[NT_G];
  #pragma unroll
  for (int i = 0; i < NT_G; ++i) { acc0[i] = 0.0f; acc1[i] = 0.0f; }

  wmma_panel2(seqbf + (size_t)(m0 + lo) * H_,
              seqbf + (size_t)(m0 + 16 + lo) * H_,
              W1bf + (size_t)lang * H_ * H_, n0, lane, acc0, acc1);

  // per-column params for this lane's N-tiles
  float biasv[NT_G], gv[NT_G], bv[NT_G];
  #pragma unroll
  for (int nt = 0; nt < NT_G; ++nt) {
    int n = n0 + nt * 16 + lo;
    biasv[nt] = b1[lang * H_ + n];
    gv[nt]    = g1[lang * H_ + n];
    bv[nt]    = beta1[lang * H_ + n];
  }
  #pragma unroll
  for (int nt = 0; nt < NT_G; ++nt)
    #pragma unroll
    for (int r = 0; r < 8; ++r) { acc0[nt][r] += biasv[nt]; acc1[nt][r] += biasv[nt]; }

  // LayerNorm over H_: per-row sum / sumsq.  Row of VGPR r: tile0 -> r+8*hi, tile1 -> 16+r+8*hi.
  float ps0[8], pq0[8], ps1[8], pq1[8];
  #pragma unroll
  for (int r = 0; r < 8; ++r) {
    float s0 = 0.f, q0 = 0.f, s1 = 0.f, q1 = 0.f;
    #pragma unroll
    for (int nt = 0; nt < NT_G; ++nt) {
      float v0 = acc0[nt][r]; s0 += v0; q0 += v0 * v0;
      float v1 = acc1[nt][r]; s1 += v1; q1 += v1 * v1;
    }
    #pragma unroll
    for (int m = 1; m < 16; m <<= 1) {     // reduce over the 16 column-lanes of each half
      s0 += __shfl_xor(s0, m, 32); q0 += __shfl_xor(q0, m, 32);
      s1 += __shfl_xor(s1, m, 32); q1 += __shfl_xor(q1, m, 32);
    }
    ps0[r] = s0; pq0[r] = q0; ps1[r] = s1; pq1[r] = q1;
  }
  __shared__ float s_sum[32][WAVES_G], s_sq[32][WAVES_G], s_mu[32], s_rs[32];
  if (lo == 0) {
    #pragma unroll
    for (int r = 0; r < 8; ++r) {
      s_sum[r + 8 * hi][wave]      = ps0[r]; s_sq[r + 8 * hi][wave]      = pq0[r];
      s_sum[16 + r + 8 * hi][wave] = ps1[r]; s_sq[16 + r + 8 * hi][wave] = pq1[r];
    }
  }
  __syncthreads();
  if (tid < 32) {
    float s = 0.f, q = 0.f;
    #pragma unroll
    for (int w = 0; w < WAVES_G; ++w) { s += s_sum[tid][w]; q += s_sq[tid][w]; }
    float mu = s * (1.0f / H_);
    float var = q * (1.0f / H_) - mu * mu;
    s_mu[tid] = mu;
    s_rs[tid] = rsqrtf(var + EPS_);
  }
  __syncthreads();
  #pragma unroll
  for (int r = 0; r < 8; ++r) {
    int row0 = r + 8 * hi, row1 = 16 + r + 8 * hi;
    float mu0 = s_mu[row0], rs0 = s_rs[row0];
    float mu1 = s_mu[row1], rs1 = s_rs[row1];
    #pragma unroll
    for (int nt = 0; nt < NT_G; ++nt) {
      float v0 = (acc0[nt][r] - mu0) * rs0 * gv[nt] + bv[nt];
      float v1 = (acc1[nt][r] - mu1) * rs1 * gv[nt] + bv[nt];
      v0 = v0 > 0.f ? v0 : 0.f;
      v1 = v1 > 0.f ? v1 : 0.f;
      h1bf[(size_t)(m0 + row0) * H_ + n0 + nt * 16 + lo] = f2bf(v0);
      h1bf[(size_t)(m0 + row1) * H_ + n0 + nt * 16 + lo] = f2bf(v1);
    }
  }
}

// ---------------- adapter layer 2 + fused masked pooling (atomic column sums) ----------------
__global__ __launch_bounds__(TPB_G) void adapter_gemm2(
    const unsigned short* __restrict__ h1bf,    // [B_*S_, H_]
    const unsigned short* __restrict__ W2bf,    // [L_, H_, H_]
    const float* __restrict__ b2, const float* __restrict__ amask,
    const int* __restrict__ lang_ids, float* __restrict__ pooled_sum) {
  const int tid = threadIdx.x, lane = tid & 31, wave = tid >> 5;
  const int hi = lane >> 4, lo = lane & 15;
  const int m0 = blockIdx.x * 32;
  const int b = m0 >> 9;
  const int s0 = m0 & (S_ - 1);
  const int lang = lang_ids[b];
  const int n0 = wave * (NT_G * 16);

  v8f acc0[NT_G], acc1[NT_G];
  #pragma unroll
  for (int i = 0; i < NT_G; ++i) { acc0[i] = 0.0f; acc1[i] = 0.0f; }

  wmma_panel2(h1bf + (size_t)(m0 + lo) * H_,
              h1bf + (size_t)(m0 + 16 + lo) * H_,
              W2bf + (size_t)lang * H_ * H_, n0, lane, acc0, acc1);

  float biasv[NT_G];
  #pragma unroll
  for (int nt = 0; nt < NT_G; ++nt) biasv[nt] = b2[lang * H_ + n0 + nt * 16 + lo];
  float mk0[8], mk1[8];
  #pragma unroll
  for (int r = 0; r < 8; ++r) {
    mk0[r] = amask[b * S_ + s0 + r + 8 * hi];
    mk1[r] = amask[b * S_ + s0 + 16 + r + 8 * hi];
  }

  // masked column sums over this tile's 32 rows, then one atomicAdd per column
  #pragma unroll
  for (int nt = 0; nt < NT_G; ++nt) {
    float cs = 0.f;
    #pragma unroll
    for (int r = 0; r < 8; ++r) {
      cs += (acc0[nt][r] + biasv[nt]) * mk0[r];
      cs += (acc1[nt][r] + biasv[nt]) * mk1[r];
    }
    cs += __shfl_xor(cs, 16, 32);          // combine the two 8-row halves (same column)
    if (hi == 0)
      atomicAdd(&pooled_sum[(size_t)b * H_ + n0 + nt * 16 + lo], cs);
  }
}

// ---------------- fusion: relu( LN( [pooled, lda] @ Wf + bf ) ) -> out ----------------
__global__ __launch_bounds__(256) void fuse_kernel(
    const float* __restrict__ pooled_sum, const float* __restrict__ msum,
    const float* __restrict__ lda, const float* __restrict__ Wf,
    const float* __restrict__ bfv, const float* __restrict__ gf,
    const float* __restrict__ betaf, float* __restrict__ out) {
  __shared__ float comb[H_ + T_];
  __shared__ float r_s[8], r_q[8], stats[2];
  const int b = blockIdx.x, tid = threadIdx.x;
  const float inv = 1.0f / msum[b];
  for (int j = tid; j < H_; j += 256) comb[j] = pooled_sum[b * H_ + j] * inv;
  for (int j = tid; j < T_; j += 256) comb[H_ + j] = lda[b * T_ + j];
  __syncthreads();

  float a0 = bfv[tid], a1 = bfv[tid + 256];
  for (int j = 0; j < H_ + T_; ++j) {
    float c = comb[j];
    a0 += c * Wf[(size_t)j * F_ + tid];
    a1 += c * Wf[(size_t)j * F_ + tid + 256];
  }
  float s = a0 + a1, q = a0 * a0 + a1 * a1;
  #pragma unroll
  for (int off = 16; off > 0; off >>= 1) {
    s += __shfl_down(s, off, 32);
    q += __shfl_down(q, off, 32);
  }
  if ((tid & 31) == 0) { r_s[tid >> 5] = s; r_q[tid >> 5] = q; }
  __syncthreads();
  if (tid == 0) {
    float S = 0.f, Q = 0.f;
    #pragma unroll
    for (int w = 0; w < 8; ++w) { S += r_s[w]; Q += r_q[w]; }
    float mu = S / F_;
    float var = Q / F_ - mu * mu;
    stats[0] = mu; stats[1] = rsqrtf(var + EPS_);
  }
  __syncthreads();
  float mu = stats[0], rs = stats[1];
  float v0 = (a0 - mu) * rs * gf[tid] + betaf[tid];
  float v1 = (a1 - mu) * rs * gf[tid + 256] + betaf[tid + 256];
  out[b * F_ + tid]       = v0 > 0.f ? v0 : 0.f;
  out[b * F_ + tid + 256] = v1 > 0.f ? v1 : 0.f;
}

// ---------------- launcher ----------------
extern "C" void kernel_launch(void* const* d_in, const int* in_sizes, int n_in,
                              void* d_out, int out_size, void* d_ws, size_t ws_size,
                              hipStream_t stream) {
  const float* seq   = (const float*)d_in[0];
  const float* amask = (const float*)d_in[1];
  const int*   lang  = (const int*)d_in[2];
  const float* lda   = (const float*)d_in[3];
  const float* W1    = (const float*)d_in[4];
  const float* b1    = (const float*)d_in[5];
  const float* g1    = (const float*)d_in[6];
  const float* be1   = (const float*)d_in[7];
  const float* W2    = (const float*)d_in[8];
  const float* b2    = (const float*)d_in[9];
  const float* Wf    = (const float*)d_in[10];
  const float* bfv   = (const float*)d_in[11];
  const float* gf    = (const float*)d_in[12];
  const float* bef   = (const float*)d_in[13];
  float* out = (float*)d_out;

  char* ws = (char*)d_ws;
  size_t off = 0;
  auto carve = [&](size_t bytes) -> char* {
    char* p = ws + off;
    off += (bytes + 255) & ~(size_t)255;
    return p;
  };
  unsigned short* seqbf  = (unsigned short*)carve((size_t)B_ * S_ * H_ * 2);
  unsigned short* h1bf   = (unsigned short*)carve((size_t)B_ * S_ * H_ * 2);
  unsigned short* W1bf   = (unsigned short*)carve((size_t)L_ * H_ * H_ * 2);
  unsigned short* W2bf   = (unsigned short*)carve((size_t)L_ * H_ * H_ * 2);
  float*          pooled = (float*)carve((size_t)B_ * H_ * 4);
  float*          msum   = (float*)carve((size_t)B_ * 4);

  const int nSeq4 = (B_ * S_ * H_) / 4;   // 4194304
  const int nW4   = (L_ * H_ * H_) / 4;   // 1310720

  cvt_f32_to_bf16_v4<<<(nSeq4 + 255) / 256, 256, 0, stream>>>(seq, seqbf, nSeq4);
  cvt_f32_to_bf16_v4<<<(nW4 + 255) / 256, 256, 0, stream>>>(W1, W1bf, nW4);
  cvt_f32_to_bf16_v4<<<(nW4 + 255) / 256, 256, 0, stream>>>(W2, W2bf, nW4);
  zero_f32<<<(B_ * H_ + 255) / 256, 256, 0, stream>>>(pooled, B_ * H_);
  mask_rowsum<<<B_, 256, 0, stream>>>(amask, msum);

  adapter_gemm1<<<(B_ * S_) / 32, TPB_G, 0, stream>>>(seqbf, W1bf, b1, g1, be1, lang, h1bf);
  adapter_gemm2<<<(B_ * S_) / 32, TPB_G, 0, stream>>>(h1bf, W2bf, b2, amask, lang, pooled);
  fuse_kernel<<<B_, 256, 0, stream>>>(pooled, msum, lda, Wf, bfv, gf, bef, out);
}
